// OTLayer_11957188952580
// MI455X (gfx1250) — matrix operand order repaired and checked by reading, compiled, not run on hardware
//
#include <hip/hip_runtime.h>
#include <hip/hip_bf16.h>

#define BATCH 32
#define HB 8
#define LL 4096
#define DD 64
#define KK 128
#define K1 129
// exp(-cost/0.05) = exp2((att-1)*20*log2(e))
#define SCALE_EXP 28.853900817779268f
#define E40 4.248354255291589e-18f   // exp(-40), trash col when mask==0

typedef __bf16 bf16;
typedef __attribute__((ext_vector_type(16))) __bf16 v16bf;
typedef __attribute__((ext_vector_type(8)))  __bf16 bf16x8;
typedef __attribute__((ext_vector_type(8)))  float  v8f;
typedef __attribute__((ext_vector_type(4)))  unsigned int u32x4;
typedef __attribute__((ext_vector_type(8)))  int i32x8;
typedef __attribute__((ext_vector_type(4)))  int i32x4;
typedef int gvec4i __attribute__((vector_size(16)));   // matches builtin param type

#if __has_builtin(__builtin_amdgcn_tensor_load_to_lds)
#define HAVE_TDM 1
#else
#define HAVE_TDM 0
#endif
#if __has_builtin(__builtin_amdgcn_global_load_async_to_lds_b128)
#define HAVE_ASYNC 1
#else
#define HAVE_ASYNC 0
#endif

union V16 { v16bf v; bf16x8 h[2]; };

__device__ __forceinline__ bf16 f2bf(float f) {
  union { float f; unsigned u; } a; a.f = f;
  unsigned u = a.u + (0x7fffu + ((a.u >> 16) & 1u));   // RNE
  union { unsigned short s; bf16 b; } r; r.s = (unsigned short)(u >> 16);
  return r.b;
}

__device__ __forceinline__ float waveRedSum(float v) {
  #pragma unroll
  for (int m = 16; m > 0; m >>= 1) v += __shfl_xor(v, m, 32);
  return v;
}

#if HAVE_ASYNC
__device__ __forceinline__ void asyncCopyB128(const bf16* gsrc, bf16* ldst) {
  __builtin_amdgcn_global_load_async_to_lds_b128(
      (__attribute__((address_space(1))) gvec4i*)gsrc,
      (__attribute__((address_space(3))) gvec4i*)ldst, 0, 0);
}
__device__ __forceinline__ void waitAsync0() {
#if __has_builtin(__builtin_amdgcn_s_wait_asynccnt)
  __builtin_amdgcn_s_wait_asynccnt(0);
#else
  asm volatile("s_wait_asynccnt 0x0" ::: "memory");
#endif
}
#endif

// ---------------- prep: feat -> bf16 + norms  (layout [N,L,H,D] -> [(nh),L,D])
__global__ __launch_bounds__(256) void k_prep_feat(const float* __restrict__ feat,
                                                   bf16* __restrict__ fbf,
                                                   float* __restrict__ fnorm) {
  int li = threadIdx.x >> 6;            // 0..3
  int d  = threadIdx.x & 63;
  long row = (long)blockIdx.x * 4 + li; // b*L + l
  int b = (int)(row >> 12);
  int l = (int)(row & 4095);
  int n = b >> 3, h = b & 7;
  float v = feat[(((long)n * LL + l) * HB + h) * DD + d];
  fbf[row * DD + d] = f2bf(v);
  float s = waveRedSum(v * v);
  __shared__ float part[8];
  if ((threadIdx.x & 31) == 0) part[threadIdx.x >> 5] = s;
  __syncthreads();
  if ((threadIdx.x & 63) == 0)
    fnorm[row] = sqrtf(part[li * 2] + part[li * 2 + 1]) + 0.5f;
}

// ---------------- prototype init (permute) + bf16 + norms
__global__ __launch_bounds__(64) void k_init_proto(const float* __restrict__ proto,
                                                   float* __restrict__ pcur,
                                                   bf16* __restrict__ pbf,
                                                   float* __restrict__ pnorm) {
  int row = blockIdx.x;                 // b*K + k
  int d = threadIdx.x;
  int b = row >> 7, k = row & 127;
  int n = b >> 3, h = b & 7;
  float v = proto[(((long)n * KK + k) * HB + h) * DD + d];
  pcur[(long)row * DD + d] = v;
  pbf[(long)row * DD + d]  = f2bf(v);
  float s = waveRedSum(v * v);
  __shared__ float part[2];
  if ((d & 31) == 0) part[d >> 5] = s;
  __syncthreads();
  if (d == 0) pnorm[row] = sqrtf(part[0] + part[1]) + 0.5f;
}

// ---------------- att GEMM: att[b,l,k] = <f_l, p_k> / (fn_l * pn_k)   (WMMA bf16)
#define LDP 136
__global__ __launch_bounds__(256) void k_att(const bf16* __restrict__ fbf,
                                             const bf16* __restrict__ pbf,
                                             const float* __restrict__ fnorm,
                                             const float* __restrict__ pnorm,
                                             float* __restrict__ att) {
  __shared__ bf16  pT[DD * LDP];        // prototypes with padded rows: pT[k][d]
  __shared__ float pn[KK];
#if HAVE_ASYNC
  __shared__ bf16  pRaw[KK * DD];
#endif
  int b  = blockIdx.y;
  int l0 = blockIdx.x * 128;
  const bf16* pb = pbf + (long)b * KK * DD;
#if HAVE_ASYNC
  // DMA prototypes into LDS (ASYNCcnt), then copy LDS->LDS into LDP-padded rows
  for (int t = threadIdx.x; t < KK * DD / 8; t += 256)
    asyncCopyB128(pb + t * 8, pRaw + t * 8);
  waitAsync0();
  __syncthreads();
  for (int idx = threadIdx.x; idx < KK * DD; idx += 256) {
    int k = idx >> 6, d = idx & 63;
    pT[k * LDP + d] = pRaw[idx];
  }
#else
  for (int idx = threadIdx.x; idx < KK * DD; idx += 256) {
    int k = idx >> 6, d = idx & 63;
    pT[k * LDP + d] = pb[idx];
  }
#endif
  if (threadIdx.x < KK) pn[threadIdx.x] = pnorm[b * KK + threadIdx.x];
  __syncthreads();

  int wave = threadIdx.x >> 5, lane = threadIdx.x & 31;
  int khalf = lane >> 4, lm = lane & 15;
  int m0 = l0 + wave * 16;
  const bf16* frow = fbf + ((long)b * LL + m0 + lm) * DD;
  int off = 8 * khalf;
  V16 a0, a1;                            // A fragments (K 0..31, 32..63)
  a0.h[0] = *(const bf16x8*)(frow + off);
  a0.h[1] = *(const bf16x8*)(frow + 16 + off);
  a1.h[0] = *(const bf16x8*)(frow + 32 + off);
  a1.h[1] = *(const bf16x8*)(frow + 48 + off);
  float fn[8];
  #pragma unroll
  for (int j = 0; j < 8; j++) fn[j] = fnorm[(long)b * LL + m0 + j + 8 * khalf];

  float* ob = att + (long)b * LL * KK;
  // software-pipelined B-fragment loads (overlap ds with wmma)
  V16 b0c, b1c, b0n, b1n;
  b0c.h[0] = *(const bf16x8*)(&pT[lm * LDP + off]);
  b0c.h[1] = *(const bf16x8*)(&pT[lm * LDP + 16 + off]);
  b1c.h[0] = *(const bf16x8*)(&pT[lm * LDP + 32 + off]);
  b1c.h[1] = *(const bf16x8*)(&pT[lm * LDP + 48 + off]);
  #pragma unroll
  for (int ct = 0; ct < 8; ct++) {
    int c0 = ct * 16;
    if (ct < 7) {
      int c1 = c0 + 16;
      b0n.h[0] = *(const bf16x8*)(&pT[(c1 + lm) * LDP + off]);
      b0n.h[1] = *(const bf16x8*)(&pT[(c1 + lm) * LDP + 16 + off]);
      b1n.h[0] = *(const bf16x8*)(&pT[(c1 + lm) * LDP + 32 + off]);
      b1n.h[1] = *(const bf16x8*)(&pT[(c1 + lm) * LDP + 48 + off]);
    }
    v8f acc = {0.f, 0.f, 0.f, 0.f, 0.f, 0.f, 0.f, 0.f};
    acc = __builtin_amdgcn_wmma_f32_16x16x32_bf16(false, a0.v, false, b0c.v, (short)0, acc, false, false);
    acc = __builtin_amdgcn_wmma_f32_16x16x32_bf16(false, a1.v, false, b1c.v, (short)0, acc, false, false);
    float pnv = pn[c0 + lm];
    #pragma unroll
    for (int j = 0; j < 8; j++) {
      int row = m0 + j + 8 * khalf;
      ob[(long)row * KK + c0 + lm] = acc[j] / (fn[j] * pnv);
    }
    if (ct < 7) { b0c = b0n; b1c = b1n; }
  }
}

// ---------------- persistent per-batch Sinkhorn + fused sim / simsum / output
__global__ __launch_bounds__(1024) void k_sinkhorn(const float* __restrict__ att,
                                                   const unsigned char* __restrict__ mask,
                                                   bf16* __restrict__ simbf,
                                                   float* __restrict__ simsum,
                                                   float* __restrict__ outF,
                                                   int finalPass) {
  __shared__ float rS[LL];
  __shared__ float cS[K1];
  __shared__ float colS[K1];
  __shared__ float red[32];
  __shared__ float sh[4];
  int b = blockIdx.x;
  int tid = threadIdx.x, lane = tid & 31, wv = tid >> 5;
  const float* A = att + (long)b * LL * KK;
  const unsigned char* M = mask + (long)(b >> 3) * LL;

  float bgp = 0.f;
  for (int l = tid; l < LL; l += 1024) bgp += (M[l] ? 0.f : 1.f);
  bgp = waveRedSum(bgp);
  if (lane == 0) red[wv] = bgp;
  __syncthreads();
  if (tid == 0) {
    float bg = 0.f;
    for (int i = 0; i < 32; i++) bg += red[i];
    sh[0] = bg / ((float)KK * (float)LL);      // v[k<K]
    sh[1] = ((float)LL - bg) / (float)LL;      // v[K]
  }
  for (int l = tid; l < LL; l += 1024) rS[l] = 1.f;
  if (tid < K1) cS[tid] = 1.f;
  __syncthreads();
  float vK = sh[0], vTr = sh[1];
  const float uVal = 1.f / (float)LL;

  float err = 1e30f;
  for (int it = 0; it < 100; ++it) {
    if (err < 0.1f) break;
    // r update: r = u / (Kmat c)
    float c0 = cS[lane], c1 = cS[lane + 32], c2 = cS[lane + 64], c3 = cS[lane + 96];
    float ctr = cS[128];
    float errp = 0.f;
    for (int l = wv; l < LL; l += 32) {
      const float* ar = A + (long)l * KK;
      __builtin_prefetch(ar + (long)32 * KK, 0, 1);
      float p = __builtin_amdgcn_exp2f((ar[lane     ] - 1.f) * SCALE_EXP) * c0
              + __builtin_amdgcn_exp2f((ar[lane + 32] - 1.f) * SCALE_EXP) * c1
              + __builtin_amdgcn_exp2f((ar[lane + 64] - 1.f) * SCALE_EXP) * c2
              + __builtin_amdgcn_exp2f((ar[lane + 96] - 1.f) * SCALE_EXP) * c3;
      if (lane == 0) p += (M[l] ? ctr : ctr * E40);
      p = waveRedSum(p);
      float rn = uVal / p;
      if (lane == 0) { errp += fabsf(rn - rS[l]); rS[l] = rn; }
    }
    if (lane == 0) red[wv] = errp;
    __syncthreads();
    if (tid == 0) { float e = 0.f; for (int i = 0; i < 32; i++) e += red[i]; sh[2] = e / (float)LL; }
    if (tid < K1) colS[tid] = 0.f;
    __syncthreads();
    err = sh[2];
    // c update: c = v / (Kmat^T r)
    float a0 = 0.f, a1 = 0.f, a2 = 0.f, a3 = 0.f, atr = 0.f;
    for (int l = wv; l < LL; l += 32) {
      const float* ar = A + (long)l * KK;
      __builtin_prefetch(ar + (long)32 * KK, 0, 1);
      float rl = rS[l];
      a0 += __builtin_amdgcn_exp2f((ar[lane     ] - 1.f) * SCALE_EXP) * rl;
      a1 += __builtin_amdgcn_exp2f((ar[lane + 32] - 1.f) * SCALE_EXP) * rl;
      a2 += __builtin_amdgcn_exp2f((ar[lane + 64] - 1.f) * SCALE_EXP) * rl;
      a3 += __builtin_amdgcn_exp2f((ar[lane + 96] - 1.f) * SCALE_EXP) * rl;
      if (lane == 0) atr += (M[l] ? 1.f : E40) * rl;
    }
    atomicAdd(&colS[lane],      a0);
    atomicAdd(&colS[lane + 32], a1);
    atomicAdd(&colS[lane + 64], a2);
    atomicAdd(&colS[lane + 96], a3);
    if (lane == 0) atomicAdd(&colS[128], atr);
    __syncthreads();
    if (tid < K1) cS[tid] = (tid < KK ? vK : vTr) / colS[tid];
    __syncthreads();
  }

  // sim = max(L * r c Kmat * att, 0)
  __syncthreads();
  if (tid < K1) colS[tid] = 0.f;
  __syncthreads();
  float cA0 = cS[lane], cA1 = cS[lane + 32], cA2 = cS[lane + 64], cA3 = cS[lane + 96];
  float ss0 = 0.f, ss1 = 0.f, ss2 = 0.f, ss3 = 0.f;
  int n = b >> 3, h = b & 7;
  for (int l = wv; l < LL; l += 32) {
    const float* ar = A + (long)l * KK;
    float rl = rS[l] * (float)LL;
    #pragma unroll
    for (int j = 0; j < 4; j++) {
      int k = lane + 32 * j;
      float cv = (j == 0 ? cA0 : j == 1 ? cA1 : j == 2 ? cA2 : cA3);
      float aa = ar[k];
      float sv = fmaxf(rl * cv * __builtin_amdgcn_exp2f((aa - 1.f) * SCALE_EXP) * aa, 0.f);
      if (finalPass) {
        outF[(((long)n * LL + l) * KK + k) * HB + h] = sv;
      } else {
        simbf[((long)b * LL + l) * KK + k] = f2bf(sv);
        if (j == 0) ss0 += sv; else if (j == 1) ss1 += sv; else if (j == 2) ss2 += sv; else ss3 += sv;
      }
    }
  }
  if (!finalPass) {
    atomicAdd(&colS[lane],      ss0);
    atomicAdd(&colS[lane + 32], ss1);
    atomicAdd(&colS[lane + 64], ss2);
    atomicAdd(&colS[lane + 96], ss3);
    __syncthreads();
    if (tid < KK) simsum[b * KK + tid] = colS[tid];
  }
}

// ---------------- proto_rec GEMM: prec[b,k,d] += sum_l sim[l,k]*f[l,d]   (WMMA bf16, split-K)
#define LTP 136
__global__ __launch_bounds__(256) void k_protorec(const bf16* __restrict__ simbf,
                                                  const bf16* __restrict__ fbf,
                                                  float* __restrict__ prec) {
  __shared__ bf16 sT[KK * LTP];          // sim^T tile: sT[k][l]
#if HAVE_TDM || HAVE_ASYNC
  __shared__ bf16 fL[128 * DD];          // staged feature chunk (TDM / async DMA)
#endif
  int b = blockIdx.y;
  int lbase0 = blockIdx.x * (LL / 4);
  int wave = threadIdx.x >> 5, lane = threadIdx.x & 31;
  int khalf = lane >> 4, lm = lane & 15;
  int m0 = wave * 16;                    // proto-row strip
  v8f zero = {0.f, 0.f, 0.f, 0.f, 0.f, 0.f, 0.f, 0.f};
  v8f acc[4];
  #pragma unroll
  for (int c = 0; c < 4; c++) acc[c] = zero;
  const bf16* S = simbf + (long)b * LL * KK;
  const bf16* F = fbf + (long)b * LL * DD;
  for (int ch = 0; ch < 8; ++ch) {
    int lb = lbase0 + ch * 128;
    // transpose-stage sim tile (VGPR path; TDM cannot transpose)
    for (int idx = threadIdx.x; idx < 128 * KK; idx += 256) {
      int lr = idx >> 7, k = idx & 127;
      sT[k * LTP + lr] = S[(long)(lb + lr) * KK + k];
    }
#if HAVE_TDM
    // Tensor Data Mover: DMA 128x64 bf16 chunk (16KB, contiguous) into LDS.
    if (threadIdx.x < 32) {
      unsigned long ga = (unsigned long)(const void*)(F + (long)lb * DD);
      unsigned ldsOff = (unsigned)(uintptr_t)(void*)fL;   // low 32 bits = LDS byte offset
      u32x4 g0;
      g0.x = 1u;                                          // count=1, user mode
      g0.y = ldsOff;                                      // lds_addr
      g0.z = (unsigned)ga;                                // global_addr[31:0]
      g0.w = (unsigned)((ga >> 32) & 0x01FFFFFFu) | 0x80000000u;  // addr[56:32] | type=2
      i32x8 g1;
      g1[0] = 0x00010000;                                 // data_size=1 (2 bytes/elem)
      g1[1] = (int)(8192u << 16);                         // tensor_dim0 = 8192 elems
      g1[2] = (int)(1u << 16);                            // tensor_dim1 = 1
      g1[3] = (int)(8192u << 16);                         // tile_dim0 = 8192 elems
      g1[4] = 0;                                          // tile_dim1/2 unused
      g1[5] = 8192;                                       // tensor_dim0_stride
      g1[6] = 0; g1[7] = 0;
      i32x4 z4 = {0, 0, 0, 0};
      i32x8 z8 = {0, 0, 0, 0, 0, 0, 0, 0};
      __builtin_amdgcn_tensor_load_to_lds(g0, g1, z4, z4, z8, 0);
      __builtin_amdgcn_s_wait_tensorcnt(0);
    }
#elif HAVE_ASYNC
    for (int t = threadIdx.x; t < 128 * DD / 8; t += 256)
      asyncCopyB128(F + (long)lb * DD + t * 8, fL + t * 8);
    waitAsync0();
#endif
    __syncthreads();
    #pragma unroll
    for (int ks = 0; ks < 4; ++ks) {
      int lofs = ks * 32 + 8 * khalf;
      V16 a;
      const bf16* arow = &sT[(m0 + lm) * LTP + lofs];
      a.h[0] = *(const bf16x8*)(arow);
      a.h[1] = *(const bf16x8*)(arow + 16);
#if HAVE_TDM || HAVE_ASYNC
      const bf16* frow = fL + (ks * 32 + 8 * khalf) * DD + lm;
#else
      const bf16* frow = F + (long)(lb + ks * 32 + 8 * khalf) * DD + lm;
#endif
      #pragma unroll
      for (int ct = 0; ct < 4; ++ct) {
        V16 bb;
        for (int s = 0; s < 8; ++s) { bb.v[s] = frow[s * DD + ct * 16]; bb.v[8 + s] = frow[(16 + s) * DD + ct * 16]; }
        acc[ct] = __builtin_amdgcn_wmma_f32_16x16x32_bf16(false, a.v, false, bb.v, (short)0, acc[ct], false, false);
      }
    }
    __syncthreads();
  }
  #pragma unroll
  for (int ct = 0; ct < 4; ++ct)
    #pragma unroll
    for (int j = 0; j < 8; j++) {
      int k = m0 + j + 8 * khalf;
      int d = ct * 16 + lm;
      atomicAdd(&prec[((long)b * KK + k) * DD + d], acc[ct][j]);
    }
}

__global__ void k_zero(float* __restrict__ p, int n) {
  int i = blockIdx.x * 256 + threadIdx.x;
  if (i < n) p[i] = 0.f;
}

// ---------------- prototype update + regenerate bf16/norms for next pass
__global__ __launch_bounds__(64) void k_proto_update(float* __restrict__ pcur,
                                                     const float* __restrict__ prec,
                                                     const float* __restrict__ simsum,
                                                     bf16* __restrict__ pbf,
                                                     float* __restrict__ pnorm) {
  int row = blockIdx.x;                  // b*K + k
  int d = threadIdx.x;
  int b = row >> 7, k = row & 127;
  float v = pcur[(long)row * DD + d] + prec[(long)row * DD + d] / (simsum[b * KK + k] + 1e-5f);
  pcur[(long)row * DD + d] = v;
  pbf[(long)row * DD + d]  = f2bf(v);
  float s = waveRedSum(v * v);
  __shared__ float part[2];
  if ((d & 31) == 0) part[d >> 5] = s;
  __syncthreads();
  if (d == 0) pnorm[row] = sqrtf(part[0] + part[1]) + 0.5f;
}

extern "C" void kernel_launch(void* const* d_in, const int* in_sizes, int n_in,
                              void* d_out, int out_size, void* d_ws, size_t ws_size,
                              hipStream_t stream) {
  (void)in_sizes; (void)n_in; (void)out_size; (void)ws_size;
  const float* feat  = (const float*)d_in[0];
  const float* proto = (const float*)d_in[1];
  const unsigned char* mask = (const unsigned char*)d_in[2];
  float* out = (float*)d_out;

  char* w = (char*)d_ws;
  size_t off = 0;
  auto alloc = [&](size_t bytes) -> void* {
    void* p = w + off; off += (bytes + 255) & ~(size_t)255; return p;
  };
  bf16*  fbf    = (bf16*) alloc((size_t)BATCH * LL * DD * 2);
  float* fnorm  = (float*)alloc((size_t)BATCH * LL * 4);
  float* pcur   = (float*)alloc((size_t)BATCH * KK * DD * 4);
  bf16*  pbf    = (bf16*) alloc((size_t)BATCH * KK * DD * 2);
  float* pnorm  = (float*)alloc((size_t)BATCH * KK * 4);
  float* att    = (float*)alloc((size_t)BATCH * LL * KK * 4);
  bf16*  simbf  = (bf16*) alloc((size_t)BATCH * LL * KK * 2);
  float* simsum = (float*)alloc((size_t)BATCH * KK * 4);
  float* prec   = (float*)alloc((size_t)BATCH * KK * DD * 4);

  k_prep_feat<<<BATCH * LL / 4, 256, 0, stream>>>(feat, fbf, fnorm);
  k_init_proto<<<BATCH * KK, 64, 0, stream>>>(proto, pcur, pbf, pnorm);

  for (int it = 0; it < 3; ++it) {
    dim3 g(LL / 128, BATCH);
    k_att<<<g, 256, 0, stream>>>(fbf, pbf, fnorm, pnorm, att);
    k_sinkhorn<<<BATCH, 1024, 0, stream>>>(att, mask, simbf, simsum, out, it == 2 ? 1 : 0);
    if (it < 2) {
      int nz = BATCH * KK * DD;
      k_zero<<<(nz + 255) / 256, 256, 0, stream>>>(prec, nz);
      dim3 g2(4, BATCH);
      k_protorec<<<g2, 256, 0, stream>>>(simbf, fbf, prec);
      k_proto_update<<<BATCH * KK, 64, 0, stream>>>(pcur, prec, simsum, pbf, pnorm);
    }
  }
}
